// MultiHeadAttention_24300924961440
// MI455X (gfx1250) — matrix operand order repaired and checked by reading, not compiled, not run on hardware
//
#include <hip/hip_runtime.h>
#include <hip/hip_bf16.h>

typedef __attribute__((ext_vector_type(16))) _Float16 v16h;
typedef __attribute__((ext_vector_type(8)))  _Float16 v8h;
typedef __attribute__((ext_vector_type(4)))  _Float16 v4h;
typedef __attribute__((ext_vector_type(8)))  float    v8f;

#define DMODEL 1024
#define NHEADS 16
#define DK     64
#define BATCH  2
#define SEQ    2048
#define MTOT   (BATCH * SEQ)      // 4096 rows

#define LDT 72    // LDS stride (halves) for 64-wide K tiles: 144B, 16B-aligned, conflict-free
#define LDP 40    // LDS stride (halves) for 32-wide tiles:    80B, 16B-aligned, conflict-free
#define L2E 1.44269504f

// ---------------------------------------------------------------------------
// CDNA5 async global->LDS copy (ASYNCcnt path, ISA 08_async_tensor §4)
// GVS addressing: mem = SGPR_base + VGPR_offset; VDST = per-lane LDS byte addr.
// ---------------------------------------------------------------------------
__device__ __forceinline__ void async_b128(unsigned lds_off, unsigned long long gbase,
                                           unsigned goff) {
  asm volatile("global_load_async_to_lds_b128 %0, %1, %2"
               :: "v"(lds_off), "v"(goff), "s"(gbase)
               : "memory");
}
__device__ __forceinline__ void wait_async0() {
  asm volatile("s_wait_asynccnt 0x0" ::: "memory");
}
// low 32 bits of a generic pointer into __shared__ = LDS byte offset (ISA §10.2)
__device__ __forceinline__ unsigned lds_off(const void* p) {
  return (unsigned)(unsigned long long)(uintptr_t)p;
}

// ---------------------------------------------------------------------------
// WMMA fragment helpers (ISA 7.12.2 layouts, wave32)
// ---------------------------------------------------------------------------
__device__ __forceinline__ v16h make_frag(const _Float16* p0, const _Float16* p1) {
  v16h r;
  *reinterpret_cast<v8h*>(&r)       = *reinterpret_cast<const v8h*>(p0);
  *(reinterpret_cast<v8h*>(&r) + 1) = *reinterpret_cast<const v8h*>(p1);
  return r;
}

// A (16x32 f16): lane holds row M=lane&15; halves K = kb..kb+7, kb+16..kb+23, kb = 8*(lane>>4)
__device__ __forceinline__ v16h load_a_frag(const _Float16* base, int stride) {
  int lane = threadIdx.x & 31;
  const _Float16* p = base + (lane & 15) * stride + ((lane >> 4) << 3);
  return make_frag(p, p + 16);
}

// B (32x16 f16) from an [N][K]-stored tile: lane holds col N=lane&15; halves K = kb..kb+7, kb+16..kb+23, kb = 8*(lane>>4)
__device__ __forceinline__ v16h load_b_frag(const _Float16* base, int stride) {
  int lane = threadIdx.x & 31;
  const _Float16* p = base + (lane & 15) * stride + ((lane >> 4) << 3);
  return make_frag(p, p + 16);
}

__device__ __forceinline__ v8f wmma16(v16h a, v16h b, v8f c) {
  return __builtin_amdgcn_wmma_f32_16x16x32_f16(false, a, false, b, (short)0, c, false, false);
}

// ---------------------------------------------------------------------------
// Kernel 0: fp32 -> f16 convert (x4 vectorized)
// ---------------------------------------------------------------------------
__global__ __launch_bounds__(256) void cvt_f32_to_f16(const float4* __restrict__ src,
                                                      v4h* __restrict__ dst, int n4) {
  int i = blockIdx.x * 256 + threadIdx.x;
  if (i < n4) {
    float4 v = src[i];
    v4h h = { (_Float16)v.x, (_Float16)v.y, (_Float16)v.z, (_Float16)v.w };
    dst[i] = h;
  }
}

// ---------------------------------------------------------------------------
// Kernel 1: fused QKV projection.  y = x @ W^T + b  (W is [N][K] row-major -> B tile direct)
// grid = (N/64, M/128, 3), block = 256 (8 waves), tile 128x64, BK=64, double-buffered
// async global->LDS pipeline.
// mode 0: Q -> [b,h,s,d] * 0.125   mode 1: K -> [b,h,s,d]   mode 2: V -> [b,h,d,s] (transposed)
// ---------------------------------------------------------------------------
__global__ __launch_bounds__(256) void qkv_gemm(
    const _Float16* __restrict__ xh,
    const _Float16* __restrict__ Wqh, const _Float16* __restrict__ Wkh,
    const _Float16* __restrict__ Wvh,
    const float* __restrict__ bq, const float* __restrict__ bk, const float* __restrict__ bv,
    _Float16* __restrict__ Qh, _Float16* __restrict__ Kh, _Float16* __restrict__ Vt) {
  __shared__ _Float16 As[2][128 * LDT];
  __shared__ _Float16 Bs[2][64 * LDT];

  const int mode = blockIdx.z;
  const _Float16* W = (mode == 0) ? Wqh : (mode == 1) ? Wkh : Wvh;
  const float* bias  = (mode == 0) ? bq  : (mode == 1) ? bk  : bv;

  const int m0 = blockIdx.y * 128;
  const int n0 = blockIdx.x * 64;
  const int t = threadIdx.x;
  const int wave = t >> 5, lane = t & 31;
  const int wm = wave >> 1, wn = wave & 1;      // 4x2 wave grid -> 32x32 per wave

  const unsigned long long xbase = (unsigned long long)(uintptr_t)xh;
  const unsigned long long wbase = (unsigned long long)(uintptr_t)W;

  // issue one 64-deep K tile (A: 1024 uint4 = 4/thread, B: 512 uint4 = 2/thread)
  auto issue_tiles = [&](int buf, int kb) {
#pragma unroll
    for (int i = 0; i < 4; ++i) {
      int idx = i * 256 + t, row = idx >> 3, seg = idx & 7;
      async_b128(lds_off(&As[buf][row * LDT + seg * 8]), xbase,
                 (unsigned)(((m0 + row) * DMODEL + kb + seg * 8) * 2));
    }
#pragma unroll
    for (int i = 0; i < 2; ++i) {
      int idx = i * 256 + t, row = idx >> 3, seg = idx & 7;
      async_b128(lds_off(&Bs[buf][row * LDT + seg * 8]), wbase,
                 (unsigned)(((n0 + row) * DMODEL + kb + seg * 8) * 2));
    }
  };

  v8f acc[2][2] = {};

  issue_tiles(0, 0);
  wait_async0();
  __syncthreads();

  int cur = 0;
  for (int kb = 0; kb < DMODEL; kb += 64) {
    int nxt = cur ^ 1;
    if (kb + 64 < DMODEL) issue_tiles(nxt, kb + 64);   // prefetch next tile (overlaps WMMAs)

#pragma unroll
    for (int kd = 0; kd < 2; ++kd) {
      v16h af[2], bf[2];
#pragma unroll
      for (int mi = 0; mi < 2; ++mi)
        af[mi] = load_a_frag(&As[cur][(wm * 32 + mi * 16) * LDT + kd * 32], LDT);
#pragma unroll
      for (int ni = 0; ni < 2; ++ni)
        bf[ni] = load_b_frag(&Bs[cur][(wn * 32 + ni * 16) * LDT + kd * 32], LDT);
#pragma unroll
      for (int mi = 0; mi < 2; ++mi)
#pragma unroll
        for (int ni = 0; ni < 2; ++ni)
          acc[mi][ni] = wmma16(af[mi], bf[ni], acc[mi][ni]);
    }

    wait_async0();       // next tile landed in LDS
    __syncthreads();     // everyone done reading cur / writing nxt
    cur = nxt;
  }

  // Epilogue: C layout -> lane = col (lane&15), VGPR i = row i + 8*(lane>>4)
  const int colk = lane & 15, rowoff = (lane >> 4) * 8;
#pragma unroll
  for (int mi = 0; mi < 2; ++mi)
#pragma unroll
    for (int ni = 0; ni < 2; ++ni) {
      int n = n0 + wn * 32 + ni * 16 + colk;
      float bn = bias[n];
      int h = n >> 6, d = n & 63;
#pragma unroll
      for (int i = 0; i < 8; ++i) {
        int m = m0 + wm * 32 + mi * 16 + rowoff + i;
        int b = m >> 11, s = m & 2047;
        float v = acc[mi][ni][i] + bn;
        if (mode == 0)
          Qh[((b * NHEADS + h) * SEQ + s) * DK + d] = (_Float16)(v * 0.125f);
        else if (mode == 1)
          Kh[((b * NHEADS + h) * SEQ + s) * DK + d] = (_Float16)v;
        else
          Vt[((b * NHEADS + h) * DK + d) * SEQ + s] = (_Float16)v;
      }
    }
}

// ---------------------------------------------------------------------------
// Kernel 2: causal flash attention, async double-buffered K/V tiles.
// grid = (B*H, SEQ/128), block = 256 (8 waves, 16 query rows per wave).
// Q pre-scaled by 1/sqrt(dk). K stored [b,h,s,d]; V stored [b,h,d,s].
// ---------------------------------------------------------------------------
__global__ __launch_bounds__(256) void attn_kernel(
    const _Float16* __restrict__ Qh, const _Float16* __restrict__ Kh,
    const _Float16* __restrict__ Vt, _Float16* __restrict__ attn) {
  __shared__ _Float16 Ks[2][32 * LDT];     // 32 keys x 64 d   (key-major = [N][K] for scores)
  __shared__ _Float16 Vs[2][64 * LDP];     // 64 d   x 32 keys (d-major   = [N][K] for P@V)
  __shared__ _Float16 Ps[8][16 * LDP];     // per-wave P staging, 16 rows x 32 keys

  const int bh = blockIdx.x;               // b*16 + h
  const int q0 = blockIdx.y * 128;         // query block base (within sequence)
  const int t = threadIdx.x;
  const int wave = t >> 5, lane = t & 31;
  const int colk = lane & 15, rowoff = (lane >> 4) * 8;
  const int qw0 = q0 + wave * 16;          // this wave's first query row
  const int qlast = qw0 + 15;

  const unsigned long long kbase = (unsigned long long)(uintptr_t)(Kh + (size_t)bh * SEQ * DK);
  const unsigned long long vbase = (unsigned long long)(uintptr_t)(Vt + (size_t)bh * DK * SEQ);

  // K tile: 256 uint4 (1/thread); V tile: 256 uint4 (1/thread)
  auto issue_tiles = [&](int buf, int kb) {
    int key = t >> 3, seg = t & 7;
    async_b128(lds_off(&Ks[buf][key * LDT + seg * 8]), kbase,
               (unsigned)(((kb + key) * DK + seg * 8) * 2));
    int d = t >> 2, sg = t & 3;
    async_b128(lds_off(&Vs[buf][d * LDP + sg * 8]), vbase,
               (unsigned)((d * SEQ + kb + sg * 8) * 2));
  };

  // Q fragments (16 rows x 64 d) straight from global, A-layout
  v16h qf[2];
#pragma unroll
  for (int kd = 0; kd < 2; ++kd) {
    const _Float16* p = Qh + ((size_t)bh * SEQ + qw0 + colk) * DK + kd * 32 + ((lane >> 4) << 3);
    qf[kd] = make_frag(p, p + 16);
  }

  v8f acc_o[4] = {};
  float m_old[8], l_sum[8];
#pragma unroll
  for (int i = 0; i < 8; ++i) { m_old[i] = -INFINITY; l_sum[i] = 0.0f; }

  const int kend = q0 + 128;               // causal upper bound for this block

  issue_tiles(0, 0);
  wait_async0();
  __syncthreads();

  int cur = 0;
  for (int kb = 0; kb < kend; kb += 32) {
    int nxt = cur ^ 1;
    if (kb + 32 < kend) issue_tiles(nxt, kb + 32);

    if (kb <= qlast) {                     // wave has at least one unmasked key here
      // scores: 16 rows x 32 keys, K-dim = 64 over two WMMA steps
      v8f s0 = {}, s1 = {};
#pragma unroll
      for (int kd = 0; kd < 2; ++kd) {
        s0 = wmma16(qf[kd], load_b_frag(&Ks[cur][0 * LDT + kd * 32], LDT), s0);
        s1 = wmma16(qf[kd], load_b_frag(&Ks[cur][16 * LDT + kd * 32], LDT), s1);
      }
      // causal mask + online softmax (row stats via 16-lane xor reductions)
#pragma unroll
      for (int i = 0; i < 8; ++i) {
        int q = qw0 + rowoff + i;
        float v0 = s0[i]; if (kb + colk > q)      v0 = -INFINITY;
        float v1 = s1[i]; if (kb + 16 + colk > q) v1 = -INFINITY;
        float rm = fmaxf(v0, v1);
#pragma unroll
        for (int mk = 1; mk < 16; mk <<= 1) rm = fmaxf(rm, __shfl_xor(rm, mk, 32));
        float mn = fmaxf(m_old[i], rm);
        float sc = exp2f((m_old[i] - mn) * L2E);
        float p0 = exp2f((v0 - mn) * L2E);
        float p1 = exp2f((v1 - mn) * L2E);
        float rs = p0 + p1;
#pragma unroll
        for (int mk = 1; mk < 16; mk <<= 1) rs += __shfl_xor(rs, mk, 32);
        l_sum[i] = l_sum[i] * sc + rs;
        m_old[i] = mn;
        acc_o[0][i] *= sc; acc_o[1][i] *= sc; acc_o[2][i] *= sc; acc_o[3][i] *= sc;
        Ps[wave][(rowoff + i) * LDP + colk]      = (_Float16)p0;
        Ps[wave][(rowoff + i) * LDP + 16 + colk] = (_Float16)p1;
      }
      // out += P(16x32) @ V(32x64); per-wave LDS is in-order, no barrier needed
      v16h pf = load_a_frag(&Ps[wave][0], LDP);
#pragma unroll
      for (int dt = 0; dt < 4; ++dt)
        acc_o[dt] = wmma16(pf, load_b_frag(&Vs[cur][dt * 16 * LDP], LDP), acc_o[dt]);
    }

    wait_async0();
    __syncthreads();
    cur = nxt;
  }

  // normalize and store as [b, s, h*64 + d] (= [B,S,1024] for the output GEMM)
  const int b = bh >> 4, h = bh & 15;
#pragma unroll
  for (int i = 0; i < 8; ++i) {
    float inv = 1.0f / l_sum[i];
    int s = qw0 + rowoff + i;
    size_t base = ((size_t)(b * SEQ + s)) * DMODEL + h * DK;
#pragma unroll
    for (int dt = 0; dt < 4; ++dt)
      attn[base + dt * 16 + colk] = (_Float16)(acc_o[dt][i] * inv);
  }
}

// ---------------------------------------------------------------------------
// Kernel 3: output projection.  out = attn @ Wo^T + bo  (fp32 out), async double-buffered
// ---------------------------------------------------------------------------
__global__ __launch_bounds__(256) void out_gemm(
    const _Float16* __restrict__ Ah, const _Float16* __restrict__ Woh,
    const float* __restrict__ bo, float* __restrict__ out) {
  __shared__ _Float16 As[2][128 * LDT];
  __shared__ _Float16 Bs[2][64 * LDT];

  const int m0 = blockIdx.y * 128;
  const int n0 = blockIdx.x * 64;
  const int t = threadIdx.x;
  const int wave = t >> 5, lane = t & 31;
  const int wm = wave >> 1, wn = wave & 1;

  const unsigned long long abase = (unsigned long long)(uintptr_t)Ah;
  const unsigned long long wbase = (unsigned long long)(uintptr_t)Woh;

  auto issue_tiles = [&](int buf, int kb) {
#pragma unroll
    for (int i = 0; i < 4; ++i) {
      int idx = i * 256 + t, row = idx >> 3, seg = idx & 7;
      async_b128(lds_off(&As[buf][row * LDT + seg * 8]), abase,
                 (unsigned)(((m0 + row) * DMODEL + kb + seg * 8) * 2));
    }
#pragma unroll
    for (int i = 0; i < 2; ++i) {
      int idx = i * 256 + t, row = idx >> 3, seg = idx & 7;
      async_b128(lds_off(&Bs[buf][row * LDT + seg * 8]), wbase,
                 (unsigned)(((n0 + row) * DMODEL + kb + seg * 8) * 2));
    }
  };

  v8f acc[2][2] = {};

  issue_tiles(0, 0);
  wait_async0();
  __syncthreads();

  int cur = 0;
  for (int kb = 0; kb < DMODEL; kb += 64) {
    int nxt = cur ^ 1;
    if (kb + 64 < DMODEL) issue_tiles(nxt, kb + 64);

#pragma unroll
    for (int kd = 0; kd < 2; ++kd) {
      v16h af[2], bf[2];
#pragma unroll
      for (int mi = 0; mi < 2; ++mi)
        af[mi] = load_a_frag(&As[cur][(wm * 32 + mi * 16) * LDT + kd * 32], LDT);
#pragma unroll
      for (int ni = 0; ni < 2; ++ni)
        bf[ni] = load_b_frag(&Bs[cur][(wn * 32 + ni * 16) * LDT + kd * 32], LDT);
#pragma unroll
      for (int mi = 0; mi < 2; ++mi)
#pragma unroll
        for (int ni = 0; ni < 2; ++ni)
          acc[mi][ni] = wmma16(af[mi], bf[ni], acc[mi][ni]);
    }

    wait_async0();
    __syncthreads();
    cur = nxt;
  }

  const int colk = lane & 15, rowoff = (lane >> 4) * 8;
#pragma unroll
  for (int mi = 0; mi < 2; ++mi)
#pragma unroll
    for (int ni = 0; ni < 2; ++ni) {
      int n = n0 + wn * 32 + ni * 16 + colk;
      float bn = bo[n];
#pragma unroll
      for (int i = 0; i < 8; ++i) {
        int m = m0 + wm * 32 + mi * 16 + rowoff + i;
        out[(size_t)m * DMODEL + n] = acc[mi][ni][i] + bn;
      }
    }
}

// ---------------------------------------------------------------------------
// Host launcher
// ---------------------------------------------------------------------------
extern "C" void kernel_launch(void* const* d_in, const int* in_sizes, int n_in,
                              void* d_out, int out_size, void* d_ws, size_t ws_size,
                              hipStream_t stream) {
  const float* x  = (const float*)d_in[0];
  const float* Wq = (const float*)d_in[1];
  const float* bq = (const float*)d_in[2];
  const float* Wk = (const float*)d_in[3];
  const float* bk = (const float*)d_in[4];
  const float* Wv = (const float*)d_in[5];
  const float* bv = (const float*)d_in[6];
  const float* Wo = (const float*)d_in[7];
  const float* bo = (const float*)d_in[8];
  float* out = (float*)d_out;

  // workspace layout (f16 halves)
  _Float16* w = (_Float16*)d_ws;
  _Float16* xh   = w; w += (size_t)MTOT * DMODEL;     // 4M
  _Float16* Wqh  = w; w += (size_t)DMODEL * DMODEL;   // 1M
  _Float16* Wkh  = w; w += (size_t)DMODEL * DMODEL;
  _Float16* Wvh  = w; w += (size_t)DMODEL * DMODEL;
  _Float16* Woh  = w; w += (size_t)DMODEL * DMODEL;
  _Float16* Qh   = w; w += (size_t)MTOT * DMODEL;     // [b,h,s,d]
  _Float16* Kh   = w; w += (size_t)MTOT * DMODEL;     // [b,h,s,d]
  _Float16* Vt   = w; w += (size_t)MTOT * DMODEL;     // [b,h,d,s]
  _Float16* attn = w; w += (size_t)MTOT * DMODEL;     // [b,s,h*d]

  // 0) convert inputs to f16
  {
    int n4 = MTOT * DMODEL / 4;                       // 1,048,576
    cvt_f32_to_f16<<<n4 / 256, 256, 0, stream>>>((const float4*)x, (v4h*)xh, n4);
    int w4 = DMODEL * DMODEL / 4;                     // 262,144
    cvt_f32_to_f16<<<w4 / 256, 256, 0, stream>>>((const float4*)Wq, (v4h*)Wqh, w4);
    cvt_f32_to_f16<<<w4 / 256, 256, 0, stream>>>((const float4*)Wk, (v4h*)Wkh, w4);
    cvt_f32_to_f16<<<w4 / 256, 256, 0, stream>>>((const float4*)Wv, (v4h*)Wvh, w4);
    cvt_f32_to_f16<<<w4 / 256, 256, 0, stream>>>((const float4*)Wo, (v4h*)Woh, w4);
  }

  // 1) QKV projections (+ scale Q, transpose V)
  qkv_gemm<<<dim3(DMODEL / 64, MTOT / 128, 3), 256, 0, stream>>>(
      xh, Wqh, Wkh, Wvh, bq, bk, bv, Qh, Kh, Vt);

  // 2) causal flash attention
  attn_kernel<<<dim3(BATCH * NHEADS, SEQ / 128), 256, 0, stream>>>(Qh, Kh, Vt, attn);

  // 3) output projection
  out_gemm<<<dim3(DMODEL / 64, MTOT / 128), 256, 0, stream>>>(attn, Woh, bo, out);
}
